// GCN_attention_11665131176121
// MI455X (gfx1250) — matrix operand order repaired and checked by reading, not compiled, not run on hardware
//
#include <hip/hip_runtime.h>
#include <hip/hip_bf16.h>
#include <math.h>

#define N     8192
#define NFEAT 256
#define NHID  64
#define NCLS  40
#define NCLSP 48   // NCLS padded to a multiple of 16 for WMMA N-tiles

typedef __attribute__((ext_vector_type(16))) __bf16 bf16x16;
typedef __attribute__((ext_vector_type(8)))  float  f32x8;

// ---------------------------------------------------------------------------
// WMMA helpers (CDNA5 wave32, V_WMMA_F32_16X16X32_BF16)
// ---------------------------------------------------------------------------
__device__ __forceinline__ f32x8 wmma_bf16(bf16x16 a, bf16x16 b, f32x8 c) {
  // (neg_a, A, neg_b, B, c_mod, C, reuse_a, reuse_b)
  return __builtin_amdgcn_wmma_f32_16x16x32_bf16(false, a, false, b, (short)0, c,
                                                 false, false);
}

// A-fragment 16x32 bf16 from row-major f32 source (adj): lane m=l&15, hi=l>>4,
// element e -> K = (e<8?0:16) + hi*8 + (e&7)  => two contiguous runs of 8.
__device__ __forceinline__ bf16x16 load_a_f32(const float* __restrict__ src,
                                              int ld, int row0, int k0, int lane) {
  const int m = lane & 15, hi = lane >> 4;
  const float* p = src + (size_t)(row0 + m) * ld + k0 + hi * 8;
  bf16x16 a;
#pragma unroll
  for (int e = 0; e < 8; ++e) a[e] = (__bf16)p[e];
#pragma unroll
  for (int e = 0; e < 8; ++e) a[e + 8] = (__bf16)p[16 + e];
  return a;
}

// Same A-fragment layout from row-major bf16 (Q)
__device__ __forceinline__ bf16x16 load_a_bf16(const __bf16* __restrict__ src,
                                               int ld, int row0, int k0, int lane) {
  const int m = lane & 15, hi = lane >> 4;
  const __bf16* p = src + (size_t)(row0 + m) * ld + k0 + hi * 8;
  bf16x16 a;
#pragma unroll
  for (int e = 0; e < 8; ++e) a[e] = p[e];
#pragma unroll
  for (int e = 0; e < 8; ++e) a[e + 8] = p[16 + e];
  return a;
}

// B-fragment 32x16 bf16: B(K,n) = src[(col0+n)*ld + k0 + K]; lane n=l&15, hi=l>>4,
// element e -> K = (e<8?0:16) + hi*8 + (e&7)  => two contiguous runs of 8.
// Works for K stored row-major (S = Q@K^T) and for V^T / XW^T / T^T stored [col][N].
__device__ __forceinline__ bf16x16 load_b_cols(const __bf16* __restrict__ src,
                                               size_t ld, int col0, int k0, int lane) {
  const int n = lane & 15, hi = lane >> 4;
  const __bf16* p = src + (size_t)(col0 + n) * ld + k0 + hi * 8;
  bf16x16 b;
#pragma unroll
  for (int e = 0; e < 8; ++e) b[e] = p[e];
#pragma unroll
  for (int e = 0; e < 8; ++e) b[e + 8] = p[16 + e];
  return b;
}

// ---------------------------------------------------------------------------
// K1a: XW^T[k][i] = (x @ W1)[i][k]   (K=256, weights staged in LDS)
// ---------------------------------------------------------------------------
__global__ void k_xw1(const float* __restrict__ x, const float* __restrict__ W1,
                      __bf16* __restrict__ XWT) {
  __shared__ float sW[NFEAT * NHID];  // 64 KB
  const int tid = threadIdx.x;
  for (int idx = tid; idx < NFEAT * NHID; idx += 256) sW[idx] = W1[idx];
  __syncthreads();
  const int wave = tid >> 5, lane = tid & 31;
  const int i = blockIdx.x * 8 + wave;
  float s0 = 0.f, s1 = 0.f;
  for (int f = 0; f < NFEAT; ++f) {
    float xv = x[(size_t)i * NFEAT + f];
    s0 = fmaf(xv, sW[f * NHID + lane], s0);
    s1 = fmaf(xv, sW[f * NHID + lane + 32], s1);
  }
  XWT[(size_t)lane * N + i]        = (__bf16)s0;
  XWT[(size_t)(lane + 32) * N + i] = (__bf16)s1;
}

// ---------------------------------------------------------------------------
// K1b: H = relu(adj @ XW + b1)   (WMMA, one wave per 16-row block)
// ---------------------------------------------------------------------------
__global__ void k_gcn_agg1(const float* __restrict__ adj, const __bf16* __restrict__ XWT,
                           const float* __restrict__ b1, float* __restrict__ H) {
  const int lane = threadIdx.x & 31;
  const int row0 = blockIdx.x * 16;
  f32x8 acc[4] = {};
  for (int kb = 0; kb < N; kb += 32) {
    if (kb + 128 < N)
      __builtin_prefetch(adj + (size_t)(row0 + (lane & 15)) * N + kb + 128, 0, 0);
    bf16x16 a = load_a_f32(adj, N, row0, kb, lane);
#pragma unroll
    for (int c = 0; c < 4; ++c) {
      bf16x16 b = load_b_cols(XWT, N, c * 16, kb, lane);
      acc[c] = wmma_bf16(a, b, acc[c]);
    }
  }
  const int n = lane & 15, hi = lane >> 4;
#pragma unroll
  for (int c = 0; c < 4; ++c)
#pragma unroll
    for (int r = 0; r < 8; ++r) {
      int row = row0 + r + 8 * hi, col = c * 16 + n;
      H[(size_t)row * NHID + col] = fmaxf(acc[c][r] + b1[col], 0.f);
    }
}

// ---------------------------------------------------------------------------
// K2: Q = H@Wq+bq (row-major bf16), K = H@Wk+bk (row-major bf16),
//     V^T = (H@Wv+bv)^T (bf16 [64][N])
// ---------------------------------------------------------------------------
__global__ void k_qkv(const float* __restrict__ H,
                      const float* __restrict__ Wq, const float* __restrict__ bq,
                      const float* __restrict__ Wk, const float* __restrict__ bk,
                      const float* __restrict__ Wv, const float* __restrict__ bv,
                      __bf16* __restrict__ Qm, __bf16* __restrict__ Km,
                      __bf16* __restrict__ VT) {
  __shared__ float sWq[NHID * NHID], sWk[NHID * NHID], sWv[NHID * NHID];  // 48 KB
  const int tid = threadIdx.x;
  for (int idx = tid; idx < NHID * NHID; idx += 256) {
    sWq[idx] = Wq[idx]; sWk[idx] = Wk[idx]; sWv[idx] = Wv[idx];
  }
  __syncthreads();
  const int wave = tid >> 5, lane = tid & 31;
  const int i = blockIdx.x * 8 + wave;
  const int o0 = lane, o1 = lane + 32;
  float q0 = bq[o0], q1 = bq[o1], k0 = bk[o0], k1 = bk[o1], v0 = bv[o0], v1 = bv[o1];
  for (int k = 0; k < NHID; ++k) {
    float h = H[(size_t)i * NHID + k];
    q0 = fmaf(h, sWq[k * NHID + o0], q0); q1 = fmaf(h, sWq[k * NHID + o1], q1);
    k0 = fmaf(h, sWk[k * NHID + o0], k0); k1 = fmaf(h, sWk[k * NHID + o1], k1);
    v0 = fmaf(h, sWv[k * NHID + o0], v0); v1 = fmaf(h, sWv[k * NHID + o1], v1);
  }
  Qm[(size_t)i * NHID + o0] = (__bf16)q0; Qm[(size_t)i * NHID + o1] = (__bf16)q1;
  Km[(size_t)i * NHID + o0] = (__bf16)k0; Km[(size_t)i * NHID + o1] = (__bf16)k1;
  VT[(size_t)o0 * N + i] = (__bf16)v0;    VT[(size_t)o1 * N + i] = (__bf16)v1;
}

// ---------------------------------------------------------------------------
// K3: fused masked attention.  A_tilde = adj * (Q@K^T); row softmax; X~ = relu(P@V).
// gcn_norm(softmax rows) is the identity (row sums == 1), folded away.
// Pass 1: per-row online (max, sumexp) over masked scores (never materialized).
// Pass 2: P = exp(s - rowmax); C-layout -> A-layout via 1KB LDS; O += P@V (WMMA).
// One wave per 16-row block.
// ---------------------------------------------------------------------------
__global__ void k_attention(const float* __restrict__ adj, const __bf16* __restrict__ Qm,
                            const __bf16* __restrict__ Km, const __bf16* __restrict__ VT,
                            float* __restrict__ Xt) {
  __shared__ __bf16 ldsP[16][32];
  const int lane = threadIdx.x & 31;
  const int n = lane & 15, hi = lane >> 4;
  const int row0 = blockIdx.x * 16;

  bf16x16 aq0 = load_a_bf16(Qm, NHID, row0, 0, lane);
  bf16x16 aq1 = load_a_bf16(Qm, NHID, row0, 32, lane);

  float rm[8], rl[8];
#pragma unroll
  for (int r = 0; r < 8; ++r) { rm[r] = -INFINITY; rl[r] = 0.f; }

  // ---- pass 1: row max & sum-of-exp of masked scores ----
  for (int jb = 0; jb < N; jb += 16) {
    if (jb + 128 < N)
      __builtin_prefetch(adj + (size_t)(row0 + (lane & 15)) * N + jb + 128, 0, 0);
    bf16x16 b0 = load_b_cols(Km, NHID, jb, 0, lane);
    bf16x16 b1 = load_b_cols(Km, NHID, jb, 32, lane);
    f32x8 s = {};
    s = wmma_bf16(aq0, b0, s);
    s = wmma_bf16(aq1, b1, s);
#pragma unroll
    for (int r = 0; r < 8; ++r) {
      float msk = adj[(size_t)(row0 + r + 8 * hi) * N + jb + n];
      float xv = msk * s[r];  // masked entries are exactly 0, as in reference
      float mn = fmaxf(rm[r], xv);
      rl[r] = rl[r] * __expf(rm[r] - mn) + __expf(xv - mn);
      rm[r] = mn;
    }
  }
  // merge (m,l) across the 16 lanes of each half-wave (all lanes end with row stats)
#pragma unroll
  for (int r = 0; r < 8; ++r) {
    float m = rm[r], l = rl[r];
    for (int off = 1; off < 16; off <<= 1) {
      float mo = __shfl_xor(m, off, 32);
      float lo = __shfl_xor(l, off, 32);
      float mn = fmaxf(m, mo);
      l = l * __expf(m - mn) + lo * __expf(mo - mn);
      m = mn;
    }
    rm[r] = m; rl[r] = l;
  }

  // ---- pass 2: O = P @ V ----
  f32x8 acc[4] = {};
  for (int jb = 0; jb < N; jb += 32) {
#pragma unroll
    for (int t = 0; t < 2; ++t) {
      const int j = jb + t * 16;
      bf16x16 b0 = load_b_cols(Km, NHID, j, 0, lane);
      bf16x16 b1 = load_b_cols(Km, NHID, j, 32, lane);
      f32x8 s = {};
      s = wmma_bf16(aq0, b0, s);
      s = wmma_bf16(aq1, b1, s);
#pragma unroll
      for (int r = 0; r < 8; ++r) {
        float msk = adj[(size_t)(row0 + r + 8 * hi) * N + j + n];
        float p = __expf(msk * s[r] - rm[r]);
        ldsP[r + 8 * hi][t * 16 + n] = (__bf16)p;  // C-layout -> row-major tile
      }
    }
    __syncthreads();
    bf16x16 pa;  // A-layout read of the 16x32 P tile
    {
      const int m = lane & 15;
#pragma unroll
      for (int e = 0; e < 8; ++e) pa[e] = ldsP[m][hi * 8 + e];
#pragma unroll
      for (int e = 0; e < 8; ++e) pa[e + 8] = ldsP[m][16 + hi * 8 + e];
    }
#pragma unroll
    for (int c = 0; c < 4; ++c) {
      bf16x16 b = load_b_cols(VT, N, c * 16, jb, lane);
      acc[c] = wmma_bf16(pa, b, acc[c]);
    }
    __syncthreads();
  }
#pragma unroll
  for (int c = 0; c < 4; ++c)
#pragma unroll
    for (int r = 0; r < 8; ++r) {
      int row = row0 + r + 8 * hi, col = c * 16 + n;
      Xt[(size_t)row * NHID + col] = fmaxf(acc[c][r] / rl[r], 0.f);
    }
}

// ---------------------------------------------------------------------------
// K4a: T^T[o][i] = (X~ @ W2)[i][o], columns padded with zeros to NCLSP=48
// ---------------------------------------------------------------------------
__global__ void k_tw2(const float* __restrict__ Xt, const float* __restrict__ W2,
                      __bf16* __restrict__ TT) {
  __shared__ float sW[NHID * NCLS];  // 10 KB
  const int tid = threadIdx.x;
  for (int idx = tid; idx < NHID * NCLS; idx += 256) sW[idx] = W2[idx];
  __syncthreads();
  const int wave = tid >> 5, lane = tid & 31;
  const int i = blockIdx.x * 8 + wave;
  const int o0 = lane;        // 0..31 (always valid)
  const int o1 = lane + 32;   // 32..63 (store pad rows up to 47)
  float s0 = 0.f, s1 = 0.f;
  for (int k = 0; k < NHID; ++k) {
    float h = Xt[(size_t)i * NHID + k];
    s0 = fmaf(h, sW[k * NCLS + o0], s0);
    if (o1 < NCLS) s1 = fmaf(h, sW[k * NCLS + o1], s1);
  }
  TT[(size_t)o0 * N + i] = (__bf16)s0;
  if (o1 < NCLSP) TT[(size_t)o1 * N + i] = (o1 < NCLS) ? (__bf16)s1 : (__bf16)0.f;
}

// ---------------------------------------------------------------------------
// K4b: z = adj @ T + b2, then row softmax over 40 classes -> out
// ---------------------------------------------------------------------------
__global__ void k_gcn_agg2(const float* __restrict__ adj, const __bf16* __restrict__ TT,
                           const float* __restrict__ b2, float* __restrict__ out) {
  const int lane = threadIdx.x & 31;
  const int row0 = blockIdx.x * 16;
  f32x8 acc[3] = {};
  for (int kb = 0; kb < N; kb += 32) {
    if (kb + 128 < N)
      __builtin_prefetch(adj + (size_t)(row0 + (lane & 15)) * N + kb + 128, 0, 0);
    bf16x16 a = load_a_f32(adj, N, row0, kb, lane);
#pragma unroll
    for (int c = 0; c < 3; ++c) {
      bf16x16 b = load_b_cols(TT, N, c * 16, kb, lane);
      acc[c] = wmma_bf16(a, b, acc[c]);
    }
  }
  const int n = lane & 15, hi = lane >> 4;
#pragma unroll
  for (int r = 0; r < 8; ++r) {
    float v[3];
    float m = -INFINITY;
#pragma unroll
    for (int c = 0; c < 3; ++c) {
      int col = c * 16 + n;
      v[c] = acc[c][r] + b2[col < NCLS ? col : 0];
      if (col < NCLS) m = fmaxf(m, v[c]);
    }
    for (int off = 1; off < 16; off <<= 1) m = fmaxf(m, __shfl_xor(m, off, 32));
    float s = 0.f;
#pragma unroll
    for (int c = 0; c < 3; ++c) {
      int col = c * 16 + n;
      v[c] = (col < NCLS) ? __expf(v[c] - m) : 0.f;
      s += v[c];
    }
    for (int off = 1; off < 16; off <<= 1) s += __shfl_xor(s, off, 32);
    const float inv = 1.f / s;
    const int row = row0 + r + 8 * hi;
#pragma unroll
    for (int c = 0; c < 3; ++c) {
      int col = c * 16 + n;
      if (col < NCLS) out[(size_t)row * NCLS + col] = v[c] * inv;
    }
  }
}

// ---------------------------------------------------------------------------
// Host side
// ---------------------------------------------------------------------------
extern "C" void kernel_launch(void* const* d_in, const int* in_sizes, int n_in,
                              void* d_out, int out_size, void* d_ws, size_t ws_size,
                              hipStream_t stream) {
  const float* adj = (const float*)d_in[0];
  const float* x   = (const float*)d_in[1];
  const float* W1  = (const float*)d_in[2];
  const float* b1  = (const float*)d_in[3];
  const float* Wq  = (const float*)d_in[4];
  const float* bq  = (const float*)d_in[5];
  const float* Wk  = (const float*)d_in[6];
  const float* bk  = (const float*)d_in[7];
  const float* Wv  = (const float*)d_in[8];
  const float* bv  = (const float*)d_in[9];
  const float* W2  = (const float*)d_in[10];
  const float* b2  = (const float*)d_in[11];
  float* out = (float*)d_out;

  char* ws = (char*)d_ws;
  size_t off = 0;
  auto alloc = [&](size_t bytes) {
    void* p = ws + off;
    off = (off + bytes + 255) & ~(size_t)255;
    return p;
  };
  __bf16* XWT = (__bf16*)alloc((size_t)NHID * N * 2);
  float*  H   = (float*)alloc((size_t)N * NHID * 4);
  __bf16* Qm  = (__bf16*)alloc((size_t)N * NHID * 2);
  __bf16* Km  = (__bf16*)alloc((size_t)N * NHID * 2);
  __bf16* VT  = (__bf16*)alloc((size_t)NHID * N * 2);
  float*  Xt  = (float*)alloc((size_t)N * NHID * 4);
  __bf16* TT  = (__bf16*)alloc((size_t)NCLSP * N * 2);
  if (off > ws_size) return;  // workspace too small (should not happen)

  k_xw1      <<<N / 8, 256, 0, stream>>>(x, W1, XWT);
  k_gcn_agg1 <<<N / 16, 32, 0, stream>>>(adj, XWT, b1, H);
  k_qkv      <<<N / 8, 256, 0, stream>>>(H, Wq, bq, Wk, bk, Wv, bv, Qm, Km, VT);
  k_attention<<<N / 16, 32, 0, stream>>>(adj, Qm, Km, VT, Xt);
  k_tw2      <<<N / 8, 256, 0, stream>>>(Xt, W2, TT);
  k_gcn_agg2 <<<N / 16, 32, 0, stream>>>(adj, TT, b2, out);
}
